// CrossAttentionLayer_66571993088151
// MI455X (gfx1250) — matrix operand order repaired and checked by reading, not compiled, not run on hardware
//
#include <hip/hip_runtime.h>
#include <hip/hip_bf16.h>

// ---- problem constants (match reference) ----
constexpr int Bn  = 16;
constexpr int Qn  = 512;
constexpr int Pn  = 2048;
constexpr int Dn  = 256;
constexpr int Hn  = 8;
constexpr int DHn = 32;

typedef __attribute__((ext_vector_type(16))) _Float16 v16h;
typedef __attribute__((ext_vector_type(8)))  _Float16 v8h;
typedef __attribute__((ext_vector_type(8)))  float    v8f;
typedef __attribute__((ext_vector_type(4)))  float    v4f;

union F16x16 { v16h v; v8h h[2]; };

// D = A(16x32 f16) * B(32x16 f16) + C(16x16 f32)
__device__ __forceinline__ v8f wmma_f16(v16h a, v16h b, v8f c) {
  return __builtin_amdgcn_wmma_f32_16x16x32_f16(
      /*neg_a=*/false, a, /*neg_b=*/false, b,
      /*c_mod=*/(short)0, c, /*reuse_a=*/false, /*reuse_b=*/false);
}

// A-matrix fragment (16x32, f16). ISA layout: lane m(0-15)/half h:
//   VGPR halves cover K = [k+8h .. k+8h+7] and [k+16+8h .. k+23+8h]
__device__ __forceinline__ v16h loadA16(const _Float16* base, int lda, int kbase) {
  const int lane = threadIdx.x & 31;
  const int m = lane & 15, hh = lane >> 4;
  const _Float16* p = base + (size_t)m * lda + kbase + 8 * hh;
  F16x16 f;
  f.h[0] = *(const v8h*)(p);
  f.h[1] = *(const v8h*)(p + 16);
  return f.v;
}

// B-matrix fragment (32x16, f16). Lane n(0-15)/half h holds column n,
//   K = [k+8h .. k+8h+7] and [k+16+8h .. k+23+8h].
__device__ __forceinline__ v16h loadB16(const _Float16* base, int ldb, int kbase) {
  const int lane = threadIdx.x & 31;
  const int n = lane & 15, hh = lane >> 4;
  const _Float16* p = base + (size_t)n * ldb + kbase + 8 * hh;
  F16x16 f;
  f.h[0] = *(const v8h*)(p);
  f.h[1] = *(const v8h*)(p + 16);
  return f.v;
}

// A fragment from f32 source, converting to f16 on the fly.
__device__ __forceinline__ v16h loadA32cvt(const float* base, int lda, int kbase) {
  const int lane = threadIdx.x & 31;
  const int m = lane & 15, hh = lane >> 4;
  const float* p = base + (size_t)m * lda + kbase + 8 * hh;
  v4f a0 = *(const v4f*)(p);
  v4f a1 = *(const v4f*)(p + 4);
  v4f a2 = *(const v4f*)(p + 16);
  v4f a3 = *(const v4f*)(p + 20);
  v16h r;
#pragma unroll
  for (int i = 0; i < 4; ++i) {
    r[i]      = (_Float16)a0[i];
    r[4 + i]  = (_Float16)a1[i];
    r[8 + i]  = (_Float16)a2[i];
    r[12 + i] = (_Float16)a3[i];
  }
  return r;
}

// ---------------- kernel 0: f32 -> f16 weight conversion ----------------
__global__ void cvt_f32_f16_kernel(const float* __restrict__ src,
                                   _Float16* __restrict__ dst, int n) {
  int i = blockIdx.x * 256 + threadIdx.x;
  if (i < n) dst[i] = (_Float16)src[i];
}

// ---------------- kernel 1: projection GEMM  out = (in @ W^T + b) * scale ---
// in: [rows, Dn] f32 ; W row-major [Dn, Dn] f16 ; out f16 [rows, Dn] or,
// if doTranspose, out[b][col][p] with row = b*Pdim + p  (V stored transposed).
__global__ __launch_bounds__(256) void proj_kernel(
    const float* __restrict__ in, const _Float16* __restrict__ w16,
    const float* __restrict__ bias, _Float16* __restrict__ out,
    int rows, int doTranspose, float scale, int Pdim) {
  const int wave = threadIdx.x >> 5;
  const int lane = threadIdx.x & 31;
  const int n = lane & 15, hh = lane >> 4;
  int id = blockIdx.x * 8 + wave;      // wave-tile id
  int nt = id & 3;                     // 4 col-tiles of 64
  int rt = id >> 2;                    // row tile (16 rows)
  if (rt * 16 >= rows) return;

  const float* aBase = in + (size_t)rt * 16 * Dn;
  v8f acc[4] = {{}, {}, {}, {}};
#pragma unroll
  for (int kb = 0; kb < Dn; kb += 32) {
    v16h a = loadA32cvt(aBase, Dn, kb);
#pragma unroll
    for (int j = 0; j < 4; ++j) {
      v16h bf = loadB16(w16 + (size_t)(nt * 64 + j * 16) * Dn, Dn, kb);
      acc[j] = wmma_f16(a, bf, acc[j]);
    }
  }
#pragma unroll
  for (int j = 0; j < 4; ++j) {
    int col = nt * 64 + j * 16 + n;
    float bc = bias[col];
#pragma unroll
    for (int r = 0; r < 8; ++r) {
      int row = rt * 16 + r + 8 * hh;
      float val = (acc[j][r] + bc) * scale;
      if (doTranspose) {
        int bb = row / Pdim, p = row % Pdim;
        out[((size_t)bb * Dn + col) * Pdim + p] = (_Float16)val;
      } else {
        out[(size_t)row * Dn + col] = (_Float16)val;
      }
    }
  }
}

// ---------------- kernel 2: fused flash attention ----------------
// one wave handles (b, h, 16-query tile); streams P in chunks of 32 keys.
__global__ __launch_bounds__(256) void attn_kernel(
    const _Float16* __restrict__ q16, const _Float16* __restrict__ k16,
    const _Float16* __restrict__ vt16, const unsigned char* __restrict__ mask,
    _Float16* __restrict__ o16) {
  __shared__ _Float16 Pl[8][16][32];   // wave-private P tiles
  const int wave = threadIdx.x >> 5;
  const int lane = threadIdx.x & 31;
  const int n = lane & 15, hh = lane >> 4;

  int id = blockIdx.x * 8 + wave;      // 4096 tiles total
  int qt = id & 31;                    // Q/16
  int h  = (id >> 5) & 7;
  int b  = id >> 8;
  const int qbase = qt * 16;

  const _Float16* qP = q16 + ((size_t)(b * Qn + qbase)) * Dn + h * DHn;
  const _Float16* kP = k16 + ((size_t)b * Pn) * Dn + h * DHn;        // stride Dn
  const _Float16* vP = vt16 + ((size_t)(b * Dn + h * DHn)) * Pn;     // stride Pn
  const unsigned char* mP = mask + ((size_t)(b * Qn + qbase)) * Pn;

  v16h qA = loadA16(qP, Dn, 0);        // scale already folded into q16

  v8f accO0 = {}, accO1 = {};
  float mrun[8], lrun[8];
#pragma unroll
  for (int r = 0; r < 8; ++r) { mrun[r] = -1e30f; lrun[r] = 0.0f; }

  for (int c0 = 0; c0 < Pn; c0 += 32) {
    if (c0 + 32 < Pn)
      __builtin_prefetch(kP + (size_t)(c0 + 32) * Dn, 0, 0);  // global_prefetch_b8

    v16h kb0 = loadB16(kP + (size_t)c0 * Dn, Dn, 0);
    v16h kb1 = loadB16(kP + (size_t)(c0 + 16) * Dn, Dn, 0);
    v8f z = {};
    v8f S0 = wmma_f16(qA, kb0, z);
    v8f S1 = wmma_f16(qA, kb1, z);

#pragma unroll
    for (int r = 0; r < 8; ++r) {
      int qrow = r + 8 * hh;           // row within tile for this lane-half
      const unsigned char* mrow = mP + (size_t)qrow * Pn + c0 + n;
      float s0 = S0[r], s1 = S1[r];
      if (mrow[0])  s0 = -1e30f;
      if (mrow[16]) s1 = -1e30f;
      // row-max across the 16 lanes of this half
      float cm = fmaxf(s0, s1);
      cm = fmaxf(cm, __shfl_xor(cm, 1, 32));
      cm = fmaxf(cm, __shfl_xor(cm, 2, 32));
      cm = fmaxf(cm, __shfl_xor(cm, 4, 32));
      cm = fmaxf(cm, __shfl_xor(cm, 8, 32));
      float mnew  = fmaxf(mrun[r], cm);
      float alpha = __expf(mrun[r] - mnew);
      mrun[r] = mnew;
      float p0 = __expf(s0 - mnew);
      float p1 = __expf(s1 - mnew);
      float rs = p0 + p1;
      rs += __shfl_xor(rs, 1, 32);
      rs += __shfl_xor(rs, 2, 32);
      rs += __shfl_xor(rs, 4, 32);
      rs += __shfl_xor(rs, 8, 32);
      lrun[r] = lrun[r] * alpha + rs;
      accO0[r] *= alpha;
      accO1[r] *= alpha;
      Pl[wave][qrow][n]      = (_Float16)p0;   // D-layout -> LDS
      Pl[wave][qrow][16 + n] = (_Float16)p1;
    }
    // reload P in A-layout from wave-private LDS (same-wave LDS is in-order)
    v16h pA  = loadA16(&Pl[wave][0][0], 32, 0);
    v16h vb0 = loadB16(vP + c0, Pn, 0);
    v16h vb1 = loadB16(vP + (size_t)16 * Pn + c0, Pn, 0);
    accO0 = wmma_f16(pA, vb0, accO0);
    accO1 = wmma_f16(pA, vb1, accO1);
  }

#pragma unroll
  for (int r = 0; r < 8; ++r) {
    int qrow = qbase + r + 8 * hh;
    float inv = 1.0f / lrun[r];        // col 0 always unmasked -> lrun > 0
    _Float16* op = o16 + ((size_t)(b * Qn + qrow)) * Dn + h * DHn + n;
    op[0]  = (_Float16)(accO0[r] * inv);
    op[16] = (_Float16)(accO1[r] * inv);
  }
}

// ---------------- kernel 3: out-proj + bias + residual + LayerNorm ----------
__global__ __launch_bounds__(128) void outln_kernel(
    const _Float16* __restrict__ o16, const _Float16* __restrict__ wo16,
    const float* __restrict__ ob, const float* __restrict__ residual,
    const float* __restrict__ ln_g, const float* __restrict__ ln_b,
    float* __restrict__ out) {
  __shared__ float tile[16][Dn + 4];
  __shared__ float mu[16], rstd[16];
  const int wave = threadIdx.x >> 5;
  const int lane = threadIdx.x & 31;
  const int n = lane & 15, hh = lane >> 4;
  const int rt = blockIdx.x;           // 512 blocks x 16 rows

  const _Float16* aBase = o16 + (size_t)rt * 16 * Dn;
  v8f acc[4] = {{}, {}, {}, {}};
#pragma unroll
  for (int kb = 0; kb < Dn; kb += 32) {
    v16h a = loadA16(aBase, Dn, kb);
#pragma unroll
    for (int j = 0; j < 4; ++j) {
      v16h bf = loadB16(wo16 + (size_t)(wave * 64 + j * 16) * Dn, Dn, kb);
      acc[j] = wmma_f16(a, bf, acc[j]);
    }
  }
#pragma unroll
  for (int j = 0; j < 4; ++j) {
    int col = wave * 64 + j * 16 + n;
    float bc = ob[col];
#pragma unroll
    for (int r = 0; r < 8; ++r) {
      int lrow = r + 8 * hh;
      int grow = rt * 16 + lrow;
      tile[lrow][col] = acc[j][r] + bc + residual[(size_t)grow * Dn + col];
    }
  }
  __syncthreads();
  if (threadIdx.x < 16) {
    int rr = threadIdx.x;
    float s = 0.0f;
    for (int c = 0; c < Dn; ++c) s += tile[rr][c];
    float m = s * (1.0f / Dn);
    float v = 0.0f;
    for (int c = 0; c < Dn; ++c) { float d = tile[rr][c] - m; v += d * d; }
    mu[rr] = m;
    rstd[rr] = rsqrtf(v * (1.0f / Dn) + 1e-5f);
  }
  __syncthreads();
  for (int i = threadIdx.x; i < 16 * Dn; i += 128) {
    int rr = i >> 8;          // / 256
    int c  = i & (Dn - 1);
    out[((size_t)rt * 16 + rr) * Dn + c] =
        (tile[rr][c] - mu[rr]) * rstd[rr] * ln_g[c] + ln_b[c];
  }
}

// ---------------- host side ----------------
extern "C" void kernel_launch(void* const* d_in, const int* in_sizes, int n_in,
                              void* d_out, int out_size, void* d_ws, size_t ws_size,
                              hipStream_t stream) {
  (void)in_sizes; (void)n_in; (void)out_size; (void)ws_size;
  const float* sources   = (const float*)d_in[0];           // [B,P,D]
  const float* queries   = (const float*)d_in[1];           // [B,Q,D]
  const unsigned char* attn_mask = (const unsigned char*)d_in[2]; // [B,Q,P] bool
  const float* in_proj_w = (const float*)d_in[3];           // [3D,D]
  const float* in_proj_b = (const float*)d_in[4];           // [3D]
  const float* out_w     = (const float*)d_in[5];           // [D,D]
  const float* out_b     = (const float*)d_in[6];           // [D]
  const float* ln_g      = (const float*)d_in[7];
  const float* ln_b      = (const float*)d_in[8];
  float* out = (float*)d_out;                               // [B,Q,D] f32

  // workspace layout (f16), ~40.5 MB total
  _Float16* w16  = (_Float16*)d_ws;                  // 3*D*D  (wq,wk,wv)
  _Float16* wo16 = w16  + 3 * Dn * Dn;               // D*D
  _Float16* q16  = wo16 + Dn * Dn;                   // B*Q*D  (pre-scaled)
  _Float16* k16  = q16  + (size_t)Bn * Qn * Dn;      // B*P*D
  _Float16* vt16 = k16  + (size_t)Bn * Pn * Dn;      // B*D*P  (transposed V)
  _Float16* o16  = vt16 + (size_t)Bn * Pn * Dn;      // B*Q*D  (attn output)

  const float qscale = 0.17677669529663687f;         // 1/sqrt(DH)

  // 0) convert weights to f16
  cvt_f32_f16_kernel<<<(3 * Dn * Dn + 255) / 256, 256, 0, stream>>>(in_proj_w, w16, 3 * Dn * Dn);
  cvt_f32_f16_kernel<<<(Dn * Dn + 255) / 256, 256, 0, stream>>>(out_w, wo16, Dn * Dn);

  // 1) projections
  {
    int rowsQ = Bn * Qn;                 // 8192
    int tilesQ = (rowsQ / 16) * 4;       // 2048 wave-tiles
    proj_kernel<<<(tilesQ + 7) / 8, 256, 0, stream>>>(
        queries, w16, in_proj_b, q16, rowsQ, 0, qscale, Pn);

    int rowsS = Bn * Pn;                 // 32768
    int tilesS = (rowsS / 16) * 4;       // 8192 wave-tiles
    proj_kernel<<<(tilesS + 7) / 8, 256, 0, stream>>>(
        sources, w16 + (size_t)Dn * Dn, in_proj_b + Dn, k16, rowsS, 0, 1.0f, Pn);
    proj_kernel<<<(tilesS + 7) / 8, 256, 0, stream>>>(
        sources, w16 + (size_t)2 * Dn * Dn, in_proj_b + 2 * Dn, vt16, rowsS, 1, 1.0f, Pn);
  }

  // 2) fused flash attention: B*H*(Q/16) = 4096 wave-tiles, 8 waves/block
  attn_kernel<<<Bn * Hn * (Qn / 16) / 8, 256, 0, stream>>>(q16, k16, vt16, attn_mask, o16);

  // 3) out-proj + residual + LayerNorm: 16 rows per block
  outln_kernel<<<(Bn * Qn) / 16, 128, 0, stream>>>(o16, wo16, out_b, queries,
                                                   ln_g, ln_b, out);
}
